// GriffinBlock_21672404976192
// MI455X (gfx1250) — matrix operand order repaired and checked by reading, not compiled, not run on hardware
//
#include <hip/hip_runtime.h>
#include <hip/hip_bf16.h>
#include <stdint.h>

#define DIM   1024
#define HEADS 16
#define HDIM  64
#define DFF   4096
#define SEQ   2048
#define BATCH 2
#define BS    (BATCH * SEQ)   // 4096 tokens

typedef __attribute__((ext_vector_type(16))) __bf16 v16bf;
typedef __attribute__((ext_vector_type(8)))  float  v8f;
typedef __attribute__((ext_vector_type(4)))  unsigned v4u;
typedef __attribute__((ext_vector_type(8)))  int    v8i;
typedef __attribute__((ext_vector_type(4)))  int    v4i;

#if defined(__has_builtin)
# if __has_builtin(__builtin_amdgcn_tensor_load_to_lds)
#  define USE_TDM 1
# endif
#endif
#ifndef USE_TDM
# define USE_TDM 0
#endif

__device__ __forceinline__ __bf16 f2bf(float f) { return (__bf16)f; }

__device__ __forceinline__ v8f wmma_bf16(v16bf a, v16bf b, v8f c) {
  return __builtin_amdgcn_wmma_f32_16x16x32_bf16(false, a, false, b, (short)0, c,
                                                 false, false);
}

// A-matrix fragment (16x32 bf16, ISA 7.12.2): lane holds row M=lane%16; per-lane
// halves are two contiguous runs: K=[8*half,8*half+8) and K=[16+8*half, +8).
__device__ __forceinline__ v16bf ldfragA(const __bf16* rowp, int hf) {
  v16bf f;
  unsigned* fu = (unsigned*)&f;
  const unsigned* p0 = (const unsigned*)(rowp + hf * 8);
  const unsigned* p1 = (const unsigned*)(rowp + 16 + hf * 8);
#pragma unroll
  for (int i = 0; i < 4; ++i) { fu[i] = p0[i]; fu[4 + i] = p1[i]; }
  return f;
}

// B-matrix fragment (32x16 bf16): lane holds N=lane%16; K in table order
// (K=[8*half,+8) then K=[16+8*half,+8)). p points at (k=0, n=n0) of a [k][n] tile with leading dim ld.
__device__ __forceinline__ v16bf ldfragB(const __bf16* p, int ld) {
  v16bf f;
  const int ln = threadIdx.x & 31, bh = ln >> 4, bn = ln & 15;
#pragma unroll
  for (int i = 0; i < 8; ++i) {
    f[i] = p[(8 * bh + i) * ld + bn];
    f[8 + i] = p[(16 + 8 * bh + i) * ld + bn];
  }
  return f;
}

__device__ __forceinline__ void st16(__bf16* dst, uint4 d) {
  uint2 a; a.x = d.x; a.y = d.y;
  uint2 b; b.x = d.z; b.y = d.w;
  ((uint2*)dst)[0] = a;
  ((uint2*)dst)[1] = b;
}

#if USE_TDM
// Issue one TDM 2D tile load (Tensor DMA, CDNA5 ISA): copies a
// tile_d1 x tile_d0 tile (elements of 2 bytes) from global (row stride
// `stride` elements) into LDS at byte address `lds`, inserting `pad_amt+1`
// DWORDs of LDS padding after every 2^(pad_int+1) DWORDs. Wave-level op
// (EXEC ignored); tracked by TENSORcnt.
__device__ __forceinline__ void tdm_load_2d(unsigned lds, uint64_t ga,
                                            unsigned td0, unsigned td1,
                                            unsigned tile_d0, unsigned tile_d1,
                                            unsigned stride,
                                            unsigned pad_int, unsigned pad_amt) {
  v4u g0;
  g0[0] = 1u;                                    // count=1 (valid descriptor)
  g0[1] = lds;                                   // lds_addr
  g0[2] = (unsigned)ga;                          // global_addr[31:0]
  g0[3] = (unsigned)(ga >> 32) | 0x80000000u;    // global_addr[56:32] | type=2
  v8i g1;
  g1[0] = (int)((1u << 16) |                     // data_size = 2 bytes
                (1u << 20) |                     // pad_enable
                (pad_int << 22) | (pad_amt << 25));
  g1[1] = (int)((td0 & 0xFFFFu) << 16);          // tensor_dim0[15:0]
  g1[2] = (int)((td0 >> 16) | ((td1 & 0xFFFFu) << 16));   // dim0 hi | dim1 lo
  g1[3] = (int)((td1 >> 16) | (tile_d0 << 16));  // dim1 hi | tile_dim0
  g1[4] = (int)(tile_d1 & 0xFFFFu);              // tile_dim1 (tile_dim2 = 0)
  g1[5] = (int)stride;                           // tensor_dim0_stride[31:0]
  g1[6] = 0;                                     // stride hi, dim1_stride lo
  g1[7] = 0;
  v4i gz = {0, 0, 0, 0};
#if __clang_major__ >= 23
  v8i gz8 = {0, 0, 0, 0, 0, 0, 0, 0};
  __builtin_amdgcn_tensor_load_to_lds(g0, g1, gz, gz, gz8, 0);
#else
  __builtin_amdgcn_tensor_load_to_lds(g0, g1, gz, gz, 0);
#endif
}
#endif

// ---------------------------------------------------------------- elementwise
__global__ __launch_bounds__(256) void cvt_bf16_kernel(const float* __restrict__ s,
                                                       __bf16* __restrict__ d,
                                                       size_t n) {
  size_t i = (size_t)blockIdx.x * 256 + threadIdx.x;
  if (i < n) d[i] = f2bf(s[i]);
}

__global__ __launch_bounds__(256) void add_kernel(const float* __restrict__ a,
                                                  const float* __restrict__ b,
                                                  float* __restrict__ c, size_t n) {
  size_t i = (size_t)blockIdx.x * 256 + threadIdx.x;
  if (i < n) c[i] = a[i] + b[i];
}

__global__ __launch_bounds__(256) void silu_mul_kernel(__bf16* __restrict__ g,
                                                       const __bf16* __restrict__ u,
                                                       size_t n) {
  size_t i = (size_t)blockIdx.x * 256 + threadIdx.x;
  if (i < n) {
    float gv = (float)g[i], uv = (float)u[i];
    float s = gv / (1.0f + __expf(-gv));
    g[i] = f2bf(s * uv);
  }
}

__global__ __launch_bounds__(256) void final_kernel(const float* __restrict__ h2,
                                                    const float* __restrict__ mlp,
                                                    const float* __restrict__ gate,
                                                    float* __restrict__ out, size_t n) {
  size_t i = (size_t)blockIdx.x * 256 + threadIdx.x;
  if (i < n) {
    float gt = gate[i & (DIM - 1)];
    float sg = 1.0f / (1.0f + __expf(-gt));
    out[i] = h2[i] + sg * mlp[i];
  }
}

// ---------------------------------------------------------------- RMSNorm
__global__ __launch_bounds__(256) void rmsnorm_kernel(const float* __restrict__ x,
                                                      const float* __restrict__ w,
                                                      __bf16* __restrict__ out) {
  const int row = blockIdx.x;
  const float* xr = x + (size_t)row * DIM;
  float ss = 0.0f;
#pragma unroll
  for (int i = 0; i < 4; ++i) {
    float v = xr[threadIdx.x + i * 256];
    ss += v * v;
  }
#pragma unroll
  for (int off = 16; off; off >>= 1) ss += __shfl_xor(ss, off, 32);
  __shared__ float red[8];
  if ((threadIdx.x & 31) == 0) red[threadIdx.x >> 5] = ss;
  __syncthreads();
  float tot = 0.0f;
#pragma unroll
  for (int i = 0; i < 8; ++i) tot += red[i];
  const float rs = rsqrtf(tot / (float)DIM + 1e-5f);
#pragma unroll
  for (int i = 0; i < 4; ++i) {
    int c = threadIdx.x + i * 256;
    out[(size_t)row * DIM + c] = f2bf(xr[c] * rs * w[c]);
  }
}

// ---------------------------------------------------------------- RoPE / layout
__global__ __launch_bounds__(256) void rope_kernel(const float* __restrict__ C,
                                                   const float* __restrict__ cs,
                                                   const float* __restrict__ sn,
                                                   __bf16* __restrict__ out,
                                                   float scale) {
  size_t idx = (size_t)blockIdx.x * 256 + threadIdx.x;
  int feat = (int)(idx & (DIM - 1));
  size_t tok = idx >> 10;
  int b = (int)(tok >> 11);
  int s = (int)(tok & (SEQ - 1));
  int hh = feat >> 6, d = feat & 63;
  float x = C[idx];
  int dro = (d < 32) ? d + 32 : d - 32;
  float xr = C[(tok << 10) + (size_t)((hh << 6) + dro)];
  float rot = (d < 32) ? -xr : xr;
  float y = (x * cs[s * HDIM + d] + rot * sn[s * HDIM + d]) * scale;
  out[(((size_t)(b * HEADS + hh)) * SEQ + s) * HDIM + d] = f2bf(y);
}

__global__ __launch_bounds__(256) void layout_bhsd_kernel(const float* __restrict__ C,
                                                          __bf16* __restrict__ out) {
  size_t idx = (size_t)blockIdx.x * 256 + threadIdx.x;
  int feat = (int)(idx & (DIM - 1));
  size_t tok = idx >> 10;
  int b = (int)(tok >> 11);
  int s = (int)(tok & (SEQ - 1));
  int hh = feat >> 6, d = feat & 63;
  out[(((size_t)(b * HEADS + hh)) * SEQ + s) * HDIM + d] = f2bf(C[idx]);
}

// ---------------------------------------------------------------- GEMM
// C[M,N] = A[M,K] * W[N,K]^T (+bias).  A,W bf16 row-major; C f32 or bf16.
#define BM 128
#define BN 64
#define BK 32
#define ALD 40   // 32 + 8 pad (halves): TDM pad_interval=16DW(enc 3), amount=4DW(enc 3)
#define BLD 72   // 64 + 8 pad (halves)

__global__ __launch_bounds__(256) void gemm_bf16_kernel(
    const __bf16* __restrict__ A, const __bf16* __restrict__ W,
    void* __restrict__ Cout, const float* __restrict__ bias,
    int M, int N, int K, int out_bf16) {
  __shared__ __bf16 sA[BM * ALD];   // [m][k] row-major
  __shared__ __bf16 sB[BK * BLD];   // [k][n] (transposed during staging)
  const int tid = threadIdx.x;
  const int lane = tid & 31;
  const int wid = tid >> 5;
  const int hf = lane >> 4;
  const int r16 = lane & 15;
  const int wm = (wid >> 1) * 32;
  const int wn = (wid & 1) * 32;
  const size_t blockM = (size_t)blockIdx.y * BM;
  const size_t blockN = (size_t)blockIdx.x * BN;
#if USE_TDM
  const unsigned sA_lds = (unsigned)(uintptr_t)&sA[0];
#endif

  v8f acc[2][2] = {};

  for (int k0 = 0; k0 < K; k0 += BK) {
#if USE_TDM
    // Tensor Data Mover stages the whole 128x32 A tile (pads rows 32->40 halves)
    if (wid == 0)
      tdm_load_2d(sA_lds,
                  (uint64_t)(uintptr_t)(A + blockM * (size_t)K + k0),
                  (unsigned)K, (unsigned)M, BK, BM, (unsigned)K,
                  /*pad_int 16DW*/3, /*pad_amt 4DW*/3);
#else
    for (int c = tid; c < (BM * BK) / 8; c += 256) {
      int row = c >> 2, col = (c & 3) * 8;
      uint4 d = *(const uint4*)(A + (blockM + row) * (size_t)K + k0 + col);
      st16(sA + row * ALD + col, d);
    }
#endif
    // stage B tile transposed: W[n][k] -> sB[k][n]; 256 chunks, one per thread
    {
      int n = tid >> 2, kc = (tid & 3) * 8;
      union { uint4 u; __bf16 h[8]; } t;
      t.u = *(const uint4*)(W + (blockN + n) * (size_t)K + k0 + kc);
#pragma unroll
      for (int i = 0; i < 8; ++i) sB[(kc + i) * BLD + n] = t.h[i];
    }
    if (k0 + BK < K) {   // L2 prefetch of next K tiles
      __builtin_prefetch(A + (blockM + (tid & 127)) * (size_t)K + k0 + BK, 0, 3);
      __builtin_prefetch(W + (blockN + (tid & 63)) * (size_t)K + k0 + BK, 0, 3);
    }
#if USE_TDM
    if (wid == 0) __builtin_amdgcn_s_wait_tensorcnt(0);
#endif
    __syncthreads();
    v16bf a0 = ldfragA(sA + (wm + r16) * ALD, hf);
    v16bf a1 = ldfragA(sA + (wm + 16 + r16) * ALD, hf);
    v16bf b0 = ldfragB(sB + wn, BLD);
    v16bf b1 = ldfragB(sB + wn + 16, BLD);
    acc[0][0] = wmma_bf16(a0, b0, acc[0][0]);
    acc[0][1] = wmma_bf16(a0, b1, acc[0][1]);
    acc[1][0] = wmma_bf16(a1, b0, acc[1][0]);
    acc[1][1] = wmma_bf16(a1, b1, acc[1][1]);
    __syncthreads();
  }
#pragma unroll
  for (int i = 0; i < 2; ++i)
#pragma unroll
    for (int j = 0; j < 2; ++j) {
      size_t col = blockN + wn + j * 16 + r16;
      float bv = bias ? bias[col] : 0.0f;
#pragma unroll
      for (int r = 0; r < 8; ++r) {
        size_t row = blockM + wm + i * 16 + hf * 8 + r;
        float v = acc[i][j][r] + bv;
        if (out_bf16) ((__bf16*)Cout)[row * N + col] = f2bf(v);
        else          ((float*)Cout)[row * N + col]  = v;
      }
    }
}

// ---------------------------------------------------------------- Flash attention
#define QB  128
#define KB  32
#define QLD 72   // 64 + 8: TDM pad_interval=32DW(enc 4), amount=4DW(enc 3)
#define KTLD 40  // 32 + 8
#define VLD 72   // 64 + 8
#define PLD 40   // 32 + 8

__global__ __launch_bounds__(256) void flash_attn_kernel(
    const __bf16* __restrict__ Q, const __bf16* __restrict__ Kk,
    const __bf16* __restrict__ V, __bf16* __restrict__ O) {
  __shared__ __bf16 sQ[QB * QLD];        // [qrow][dim]
  __shared__ __bf16 sKt[HDIM * KTLD];    // [dim][key]   (B tile for Q*K^T)
  __shared__ __bf16 sV[KB * VLD];        // [key][dim]   (B tile for P*V)
  __shared__ __bf16 sP[8 * 16 * PLD];    // per-wave P (16x32) staging

  const int tid = threadIdx.x, lane = tid & 31, wid = tid >> 5;
  const int hf = lane >> 4, r16 = lane & 15;
  const int bh = blockIdx.y;
  const int qb = blockIdx.x;
  const int q0 = qb * QB;
  const size_t base = (size_t)bh * SEQ * HDIM;

#if USE_TDM
  if (wid == 0) {
    tdm_load_2d((unsigned)(uintptr_t)&sQ[0],
                (uint64_t)(uintptr_t)(Q + base + (size_t)q0 * HDIM),
                HDIM, SEQ, HDIM, QB, HDIM, /*32DW*/4, /*4DW*/3);
    __builtin_amdgcn_s_wait_tensorcnt(0);
  }
#else
  for (int c = tid; c < (QB * HDIM) / 8; c += 256) {
    int row = c >> 3, col = (c & 7) * 8;
    uint4 d = *(const uint4*)(Q + base + (size_t)(q0 + row) * HDIM + col);
    st16(sQ + row * QLD + col, d);
  }
#endif
  __syncthreads();
  v16bf qf0 = ldfragA(sQ + (wid * 16 + r16) * QLD, hf);        // dims 0..31
  v16bf qf1 = ldfragA(sQ + (wid * 16 + r16) * QLD + 32, hf);   // dims 32..63

  v8f o[4] = {};
  float mrun[8], lrun[8];
#pragma unroll
  for (int r = 0; r < 8; ++r) { mrun[r] = -1e30f; lrun[r] = 0.0f; }

  const int ktiles = (qb + 1) * (QB / KB);
  for (int kt = 0; kt < ktiles; ++kt) {
    const int k0 = kt * KB;
    __syncthreads();
#if USE_TDM
    if (wid == 0)   // V tile via TDM, overlapped with manual K^T staging below
      tdm_load_2d((unsigned)(uintptr_t)&sV[0],
                  (uint64_t)(uintptr_t)(V + base + (size_t)k0 * HDIM),
                  HDIM, SEQ, HDIM, KB, HDIM, /*32DW*/4, /*4DW*/3);
#endif
    {
      int key = tid >> 3, col = (tid & 7) * 8;
      union { uint4 u; __bf16 h[8]; } t;
      t.u = *(const uint4*)(Kk + base + (size_t)(k0 + key) * HDIM + col);
#pragma unroll
      for (int i = 0; i < 8; ++i) sKt[(col + i) * KTLD + key] = t.h[i];  // transpose
#if !USE_TDM
      uint4 dv = *(const uint4*)(V + base + (size_t)(k0 + key) * HDIM + col);
      st16(sV + key * VLD + col, dv);
#endif
    }
#if USE_TDM
    if (wid == 0) __builtin_amdgcn_s_wait_tensorcnt(0);
#endif
    __syncthreads();

    // S = Q * K^T
    v8f s0 = {}, s1 = {};
    s0 = wmma_bf16(qf0, ldfragB(sKt + 0, KTLD), s0);
    s0 = wmma_bf16(qf1, ldfragB(sKt + 32 * KTLD + 0, KTLD), s0);
    s1 = wmma_bf16(qf0, ldfragB(sKt + 16, KTLD), s1);
    s1 = wmma_bf16(qf1, ldfragB(sKt + 32 * KTLD + 16, KTLD), s1);

    const int qrow_base = q0 + wid * 16 + hf * 8;
    const int key0 = k0 + r16, key1 = key0 + 16;
#pragma unroll
    for (int r = 0; r < 8; ++r) {
      const int qr = qrow_base + r;
      float a0 = s0[r] + ((key0 > qr) ? -1e4f : 0.0f);
      float a1 = s1[r] + ((key1 > qr) ? -1e4f : 0.0f);
      float mx = fmaxf(a0, a1);
#pragma unroll
      for (int off = 8; off; off >>= 1) mx = fmaxf(mx, __shfl_xor(mx, off, 32));
      float mnew = fmaxf(mrun[r], mx);
      float alpha = __expf(mrun[r] - mnew);
      float p0 = __expf(a0 - mnew);
      float p1 = __expf(a1 - mnew);
      float rs = p0 + p1;
#pragma unroll
      for (int off = 8; off; off >>= 1) rs += __shfl_xor(rs, off, 32);
      lrun[r] = lrun[r] * alpha + rs;
      mrun[r] = mnew;
      o[0][r] *= alpha; o[1][r] *= alpha; o[2][r] *= alpha; o[3][r] *= alpha;
      __bf16* prow = sP + (wid * 16 + hf * 8 + r) * PLD;
      prow[r16]      = f2bf(p0);
      prow[16 + r16] = f2bf(p1);
    }
    asm volatile("s_wait_dscnt 0" ::: "memory");   // wave-private P store->load
    v16bf pf = ldfragA(sP + (wid * 16 + r16) * PLD, hf);
#pragma unroll
    for (int j = 0; j < 4; ++j)
      o[j] = wmma_bf16(pf, ldfragB(sV + j * 16, VLD), o[j]);
  }

  const int b = bh >> 4, hh = bh & 15;
#pragma unroll
  for (int j = 0; j < 4; ++j)
#pragma unroll
    for (int r = 0; r < 8; ++r) {
      int srow = q0 + wid * 16 + hf * 8 + r;
      int d = j * 16 + r16;
      O[((size_t)(b * SEQ + srow)) * DIM + hh * HDIM + d] = f2bf(o[j][r] / lrun[r]);
    }
}

// ---------------------------------------------------------------- host
extern "C" void kernel_launch(void* const* d_in, const int* in_sizes, int n_in,
                              void* d_out, int out_size, void* d_ws, size_t ws_size,
                              hipStream_t stream) {
  (void)in_sizes; (void)n_in; (void)out_size; (void)ws_size;
  const float* h      = (const float*)d_in[0];
  const float* cosT   = (const float*)d_in[1];
  const float* sinT   = (const float*)d_in[2];
  const float* n1w    = (const float*)d_in[3];
  const float* wq     = (const float*)d_in[4];
  const float* wk     = (const float*)d_in[5];
  const float* wv     = (const float*)d_in[6];
  const float* wo     = (const float*)d_in[7];
  const float* n2w    = (const float*)d_in[8];
  const float* gate_w = (const float*)d_in[9];
  const float* gate_b = (const float*)d_in[10];
  const float* up_w   = (const float*)d_in[11];
  const float* up_b   = (const float*)d_in[12];
  const float* down_w = (const float*)d_in[13];
  const float* down_b = (const float*)d_in[14];
  const float* gatev  = (const float*)d_in[15];

  char* ws = (char*)d_ws;
  size_t off = 0;
  auto alloc = [&](size_t bytes) -> void* {
    void* p = ws + off;
    off = (off + bytes + 255) & ~(size_t)255;
    return p;
  };
  __bf16* wq_bf = (__bf16*)alloc((size_t)DIM * DIM * 2);
  __bf16* wk_bf = (__bf16*)alloc((size_t)DIM * DIM * 2);
  __bf16* wv_bf = (__bf16*)alloc((size_t)DIM * DIM * 2);
  __bf16* wo_bf = (__bf16*)alloc((size_t)DIM * DIM * 2);
  __bf16* gw_bf = (__bf16*)alloc((size_t)DFF * DIM * 2);
  __bf16* uw_bf = (__bf16*)alloc((size_t)DFF * DIM * 2);
  __bf16* dw_bf = (__bf16*)alloc((size_t)DIM * DFF * 2);
  __bf16* x_bf  = (__bf16*)alloc((size_t)BS * DIM * 2);
  float*  Cf    = (float*) alloc((size_t)BS * DIM * 4);
  __bf16* q_bf  = (__bf16*)alloc((size_t)BS * DIM * 2);
  __bf16* k_bf  = (__bf16*)alloc((size_t)BS * DIM * 2);
  __bf16* v_bf  = (__bf16*)alloc((size_t)BS * DIM * 2);
  __bf16* o_bf  = (__bf16*)alloc((size_t)BS * DIM * 2);
  float*  h2    = (float*) alloc((size_t)BS * DIM * 4);
  __bf16* g_bf  = (__bf16*)alloc((size_t)BS * DFF * 2);
  __bf16* u_bf  = (__bf16*)alloc((size_t)BS * DFF * 2);

  auto cvt = [&](const float* s, __bf16* d, size_t n) {
    cvt_bf16_kernel<<<dim3((unsigned)((n + 255) / 256)), dim3(256), 0, stream>>>(s, d, n);
  };
  auto gemm = [&](const __bf16* A, const __bf16* W, void* C, const float* bias,
                  int M, int N, int K, int obf) {
    gemm_bf16_kernel<<<dim3(N / BN, M / BM), dim3(256), 0, stream>>>(A, W, C, bias,
                                                                     M, N, K, obf);
  };

  cvt(wq, wq_bf, (size_t)DIM * DIM);
  cvt(wk, wk_bf, (size_t)DIM * DIM);
  cvt(wv, wv_bf, (size_t)DIM * DIM);
  cvt(wo, wo_bf, (size_t)DIM * DIM);
  cvt(gate_w, gw_bf, (size_t)DFF * DIM);
  cvt(up_w,   uw_bf, (size_t)DFF * DIM);
  cvt(down_w, dw_bf, (size_t)DIM * DFF);

  const unsigned nBD = (unsigned)(((size_t)BS * DIM) / 256);
  const unsigned nBF = (unsigned)(((size_t)BS * DFF) / 256);

  rmsnorm_kernel<<<dim3(BS), dim3(256), 0, stream>>>(h, n1w, x_bf);
  gemm(x_bf, wq_bf, Cf, nullptr, BS, DIM, DIM, 0);
  rope_kernel<<<dim3(nBD), dim3(256), 0, stream>>>(Cf, cosT, sinT, q_bf, 0.125f);
  gemm(x_bf, wk_bf, Cf, nullptr, BS, DIM, DIM, 0);
  rope_kernel<<<dim3(nBD), dim3(256), 0, stream>>>(Cf, cosT, sinT, k_bf, 1.0f);
  gemm(x_bf, wv_bf, Cf, nullptr, BS, DIM, DIM, 0);
  layout_bhsd_kernel<<<dim3(nBD), dim3(256), 0, stream>>>(Cf, v_bf);
  flash_attn_kernel<<<dim3(SEQ / QB, BATCH * HEADS), dim3(256), 0, stream>>>(
      q_bf, k_bf, v_bf, o_bf);
  gemm(o_bf, wo_bf, Cf, nullptr, BS, DIM, DIM, 0);
  add_kernel<<<dim3(nBD), dim3(256), 0, stream>>>(h, Cf, h2, (size_t)BS * DIM);

  rmsnorm_kernel<<<dim3(BS), dim3(256), 0, stream>>>(h2, n2w, x_bf);
  gemm(x_bf, gw_bf, g_bf, gate_b, BS, DFF, DIM, 1);
  gemm(x_bf, uw_bf, u_bf, up_b,   BS, DFF, DIM, 1);
  silu_mul_kernel<<<dim3(nBF), dim3(256), 0, stream>>>(g_bf, u_bf, (size_t)BS * DFF);
  gemm(g_bf, dw_bf, Cf, down_b, BS, DIM, DFF, 0);
  final_kernel<<<dim3(nBD), dim3(256), 0, stream>>>(h2, Cf, gatev, (float*)d_out,
                                                    (size_t)BS * DIM);
}
